// SelfAttention_20753281974997
// MI455X (gfx1250) — matrix operand order repaired and checked by reading, not compiled, not run on hardware
//
#include <hip/hip_runtime.h>
#include <math.h>

// Self-attention fwd on MI455X (gfx1250): B=4, T=2048, D=2048, H=16, HD=128.
// bf16 WMMA (v_wmma_f32_16x16x32_bf16) + Tensor Data Mover (tensor_load_to_lds)
// with double-buffered TDM pipeline in the GEMMs, flash-attention core.
//  k0: fp32 -> bf16 conversion of x / w_qkv / w_o (vectorized)
//  k1: QKV GEMM (TDM-fed)    -> Q,K,V bf16 (B,H,T,HD), Q pre-scaled by 1/sqrt(HD)
//  k2: flash attention (TDM-fed K tiles, streaming softmax) -> Y bf16
//  k3: output GEMM (TDM-fed) -> fp32 d_out
// Workspace: xb/y(alias) 32MB + wqkv 24MB + wo 8MB + Q/K/V 96MB = 160MB.

#define B_  4
#define T_  2048
#define D_  2048
#define H_  16
#define HD_ 128
#define NEG_BIG (-1e30f)

typedef __attribute__((ext_vector_type(16))) __bf16  bf16x16;
typedef __attribute__((ext_vector_type(8)))  float   f32x8;
typedef __attribute__((ext_vector_type(4)))  unsigned u32x4;
typedef __attribute__((ext_vector_type(8)))  int     i32x8;
typedef __attribute__((ext_vector_type(4)))  int     i32x4;

// D# group1 word0 presets: data_size=2B (bits17:16=1), pad_enable (bit20),
// pad_interval (bits24:22, 8B units = 1<<code), pad_amount (bits31:25, DWORDs = code+1)
#define TDM_CFG_GEMM  0x0ED10000u  // pad 8 DW (16 bf16) after 8 units (32 bf16 row)
#define TDM_CFG_FLASH 0x07510000u  // pad 4 DW ( 8 bf16) after 32 units (128 bf16 row)

#define LDS_OFF(p) ((unsigned)(unsigned long long)(const void*)(p))

__device__ __forceinline__ __bf16 f2bf(float f) {
  union { float f; unsigned u; } cv; cv.f = f;
  unsigned r = cv.u + 0x7FFFu + ((cv.u >> 16) & 1u);   // RNE
  unsigned short h = (unsigned short)(r >> 16);
  return __builtin_bit_cast(__bf16, h);
}

__device__ __forceinline__ unsigned pack_bf2(float a, float b) {
  union { float f; unsigned u; } x, y; x.f = a; y.f = b;
  unsigned lo = (x.u + 0x7FFFu + ((x.u >> 16) & 1u)) >> 16;
  unsigned hi = (y.u + 0x7FFFu + ((y.u >> 16) & 1u)) & 0xFFFF0000u;
  return lo | hi;
}

// A-fragment (16x32 MxK bf16): lane = half*16 + m; elems 0..15 = K[half*16+i].
// p = row base + half*16 (16B aligned).
__device__ __forceinline__ bf16x16 ld_fragA(const __bf16* p) {
  bf16x16 f;
#pragma unroll
  for (int i = 0; i < 16; ++i) f[i] = p[i];
  return f;
}

// B-fragment (32x16 KxN bf16): lane = half*16 + n; elems 0..15 = K[half*16+i].
__device__ __forceinline__ bf16x16 ld_fragB(const __bf16* p) {
  bf16x16 f;
#pragma unroll
  for (int i = 0; i < 16; ++i) f[i] = p[i];
  return f;
}

__device__ __forceinline__ f32x8 wmma_bf16(bf16x16 a, bf16x16 b, f32x8 c) {
  return __builtin_amdgcn_wmma_f32_16x16x32_bf16(false, a, false, b, (short)0, c,
                                                 false, false);
}

// TDM: 2-D bf16 tile (tile_d1 rows x tile_d0 elems) global -> LDS, row stride
// d0_stride elems, LDS padding per cfg. Issue from one wave only.
// clang-23 6-arg form: (g0, g1, g2, g3, extra, cpol).
__device__ __forceinline__ void tdm_load_2d(unsigned lds_off, const void* gsrc,
                                            unsigned tensor_d0, unsigned tensor_d1,
                                            unsigned tile_d0, unsigned tile_d1,
                                            unsigned d0_stride, unsigned cfg) {
  unsigned long long ga = (unsigned long long)gsrc;
  u32x4 g0;
  g0[0] = 1u;                                             // count=1, user mode
  g0[1] = lds_off;                                        // lds_addr
  g0[2] = (unsigned)ga;                                   // global_addr[31:0]
  g0[3] = (unsigned)((ga >> 32) & 0x01FFFFFFull) | (2u << 30);  // [56:32] | type=2
  i32x8 g1;
  g1[0] = (int)cfg;                                       // mask=0|data_size|pad
  g1[1] = (int)(tensor_d0 << 16);                         // tensor_dim0[15:0]
  g1[2] = (int)((tensor_d0 >> 16) | (tensor_d1 << 16));   // d0[31:16] | d1[15:0]
  g1[3] = (int)((tensor_d1 >> 16) | (tile_d0 << 16));     // d1[31:16] | tile_d0
  g1[4] = (int)(tile_d1 & 0xFFFFu);                       // tile_d1 (tile_d2=0)
  g1[5] = (int)d0_stride;                                 // dim0_stride[31:0]
  g1[6] = 0;
  g1[7] = 0;
  i32x4 gz = {0, 0, 0, 0};                                // groups 2/3: 2-D only
  i32x8 gx = {0, 0, 0, 0, 0, 0, 0, 0};
  __builtin_amdgcn_tensor_load_to_lds(g0, g1, gz, gz, gx, 0);
}

// ---------------- fp32 -> bf16 conversion (vectorized) -------------------------
__global__ __launch_bounds__(256) void cvt_bf16(const float* __restrict__ src,
                                                __bf16* __restrict__ dst, int n4) {
  int i = blockIdx.x * 256 + threadIdx.x;
  if (i < n4) {
    float4 f = ((const float4*)src)[i];
    uint2 p;
    p.x = pack_bf2(f.x, f.y);
    p.y = pack_bf2(f.z, f.w);
    ((uint2*)dst)[i] = p;
  }
}

// ---------------- QKV projection: (8192,2048) x (2048,6144) -> Q,K,V bf16 ------
__global__ __launch_bounds__(256) void qkv_gemm(const __bf16* __restrict__ xb,
                                                const __bf16* __restrict__ wb,
                                                __bf16* __restrict__ q,
                                                __bf16* __restrict__ k,
                                                __bf16* __restrict__ v) {
  const int K = D_;
  __shared__ __bf16 As[2][128][48];   // TDM fills [m][32] rows + 16-elem pad
  __shared__ __bf16 Bs[2][128][48];   // [n][k]
  const int tid  = threadIdx.x;
  const int lane = tid & 31, wid = tid >> 5;
  const int half = lane >> 4, ln = lane & 15;
  const int wm = wid & 3, wn = wid >> 2;               // 4(M) x 2(N) waves
  const int m0 = blockIdx.y * 128, n0 = blockIdx.x * 128;
  const int waveRow = wm * 32, waveCol = wn * 64;
  const int NK = K / 32;

  f32x8 acc[2][4] = {};

  if (wid == 0) {   // prologue: stream first tile pair
    tdm_load_2d(LDS_OFF(&As[0][0][0]), xb + (size_t)m0 * K, K, B_ * T_, 32, 128, K, TDM_CFG_GEMM);
    tdm_load_2d(LDS_OFF(&Bs[0][0][0]), wb + (size_t)n0 * K, K, 3 * D_, 32, 128, K, TDM_CFG_GEMM);
  }
  for (int kt = 0; kt < NK; ++kt) {
    const int buf = kt & 1;
    if (wid == 0) {
      if (kt + 1 < NK) {   // prefetch next pair, then wait for current pair
        const int k1 = (kt + 1) * 32;
        tdm_load_2d(LDS_OFF(&As[buf ^ 1][0][0]), xb + (size_t)m0 * K + k1, K, B_ * T_, 32, 128, K, TDM_CFG_GEMM);
        tdm_load_2d(LDS_OFF(&Bs[buf ^ 1][0][0]), wb + (size_t)n0 * K + k1, K, 3 * D_, 32, 128, K, TDM_CFG_GEMM);
        __builtin_amdgcn_s_wait_tensorcnt(2);
      } else {
        __builtin_amdgcn_s_wait_tensorcnt(0);
      }
    }
    __syncthreads();    // current tiles resident
    bf16x16 af[2], bfr[4];
#pragma unroll
    for (int mi = 0; mi < 2; ++mi)
      af[mi] = ld_fragA(&As[buf][waveRow + mi * 16 + ln][half * 16]);
#pragma unroll
    for (int ni = 0; ni < 4; ++ni)
      bfr[ni] = ld_fragB(&Bs[buf][waveCol + ni * 16 + ln][half * 16]);
#pragma unroll
    for (int mi = 0; mi < 2; ++mi)
#pragma unroll
      for (int ni = 0; ni < 4; ++ni)
        acc[mi][ni] = wmma_bf16(af[mi], bfr[ni], acc[mi][ni]);
    __syncthreads();    // tiles consumed; next prefetch may overwrite
  }

  const float qscale = 0.08838834764831845f;   // 1/sqrt(HD)
#pragma unroll
  for (int mi = 0; mi < 2; ++mi) {
#pragma unroll
    for (int ni = 0; ni < 4; ++ni) {
      int col   = n0 + waveCol + ni * 16 + ln;   // 0..6143
      int which = col >> 11;                     // 0=q 1=k 2=v
      int dcol  = col & (D_ - 1);
      int h     = dcol >> 7;
      int hd    = dcol & (HD_ - 1);
      __bf16* dst = (which == 0) ? q : (which == 1) ? k : v;
      float sc = (which == 0) ? qscale : 1.0f;
#pragma unroll
      for (int r = 0; r < 8; ++r) {
        int row = m0 + waveRow + mi * 16 + half * 8 + r;
        int b = row >> 11, t = row & (T_ - 1);
        dst[((size_t)(b * H_ + h) * T_ + t) * HD_ + hd] = f2bf(acc[mi][ni][r] * sc);
      }
    }
  }
}

// ---------------- Flash attention: one (b,h,128 q-rows) per block --------------
__global__ __launch_bounds__(256) void flash_attn(const __bf16* __restrict__ q,
                                                  const __bf16* __restrict__ k,
                                                  const __bf16* __restrict__ v,
                                                  const int* __restrict__ mask,
                                                  __bf16* __restrict__ y) {
  __shared__ __bf16 Ks[64][136];       // [key][hd], TDM pad 8
  __shared__ __bf16 Vs[128][72];       // [hd][key] transposed, pad 8
  __shared__ __bf16 Ps[8][16][72];     // per-wave P (16 q-rows x 64 keys), pad 8
  __shared__ int    Msk[64];

  const int tid  = threadIdx.x;
  const int lane = tid & 31, wid = tid >> 5;
  const int half = lane >> 4, ln = lane & 15;
  const int qt = blockIdx.x, hh = blockIdx.y, bb = blockIdx.z;
  const size_t headBase = (size_t)(bb * H_ + hh) * T_ * HD_;

  // Preload this wave's 16 Q rows (full HD=128) as 4 A-fragments.
  const __bf16* qbase = q + headBase + (size_t)(qt * 128 + wid * 16) * HD_;
  bf16x16 qf[4];
#pragma unroll
  for (int kk = 0; kk < 4; ++kk)
    qf[kk] = ld_fragA(qbase + (size_t)ln * HD_ + kk * 32 + half * 16);

  f32x8 o[8] = {};
  float mrow[8], lrow[8];
#pragma unroll
  for (int r = 0; r < 8; ++r) { mrow[r] = NEG_BIG; lrow[r] = 0.0f; }

  for (int kt = 0; kt < T_ / 64; ++kt) {
    const __bf16* ksrc = k + headBase + (size_t)(kt * 64) * HD_;
    const __bf16* vsrc = v + headBase + (size_t)(kt * 64) * HD_;

    // K tile via TDM (64 rows x 128 bf16, LDS pad 8/row -> Ks[64][136])
    if (wid == 0)
      tdm_load_2d(LDS_OFF(&Ks[0][0]), ksrc, HD_, T_, HD_, 64, HD_, TDM_CFG_FLASH);

    // V tile: vectorized load + transpose scatter into Vs[hd][key]
    for (int i = tid; i < 64 * 128 / 8; i += 256) {
      int r = i >> 4;            // key
      int c = (i & 15) * 8;      // hd base
      uint4 d = *(const uint4*)(vsrc + (size_t)r * HD_ + c);
      const __bf16* e = (const __bf16*)&d;
#pragma unroll
      for (int j = 0; j < 8; ++j) Vs[c + j][r] = e[j];
    }
    if (tid < 64) Msk[tid] = mask[bb * T_ + kt * 64 + tid];
    if (wid == 0) __builtin_amdgcn_s_wait_tensorcnt(0);
    __syncthreads();

    // S = Q * K^T (16 x 64 per wave)
    f32x8 s[4] = {};
#pragma unroll
    for (int ni = 0; ni < 4; ++ni)
#pragma unroll
      for (int kk = 0; kk < 4; ++kk) {
        bf16x16 bfr = ld_fragB(&Ks[ni * 16 + ln][kk * 32 + half * 16]);
        s[ni] = wmma_bf16(qf[kk], bfr, s[ni]);
      }

    // mask + per-row max
    float vmax[8];
#pragma unroll
    for (int r = 0; r < 8; ++r) vmax[r] = NEG_BIG;
#pragma unroll
    for (int ni = 0; ni < 4; ++ni) {
      bool live = Msk[ni * 16 + ln] != 0;
#pragma unroll
      for (int r = 0; r < 8; ++r) {
        float sv = live ? s[ni][r] : NEG_BIG;
        s[ni][r] = sv;
        vmax[r] = fmaxf(vmax[r], sv);
      }
    }
#pragma unroll
    for (int off = 1; off < 16; off <<= 1)
#pragma unroll
      for (int r = 0; r < 8; ++r)
        vmax[r] = fmaxf(vmax[r], __shfl_xor(vmax[r], off, 32));

    float corr[8];
#pragma unroll
    for (int r = 0; r < 8; ++r) {
      float mn = fmaxf(mrow[r], vmax[r]);
      corr[r] = __expf(mrow[r] - mn);
      mrow[r] = mn;
    }
    float rsum[8];
#pragma unroll
    for (int r = 0; r < 8; ++r) rsum[r] = 0.0f;
#pragma unroll
    for (int ni = 0; ni < 4; ++ni)
#pragma unroll
      for (int r = 0; r < 8; ++r) {
        float p = __expf(s[ni][r] - mrow[r]);
        s[ni][r] = p;
        rsum[r] += p;
      }
#pragma unroll
    for (int r = 0; r < 8; ++r) lrow[r] = lrow[r] * corr[r] + rsum[r];
#pragma unroll
    for (int f = 0; f < 8; ++f)
#pragma unroll
      for (int r = 0; r < 8; ++r) o[f][r] *= corr[r];

    // P: C-layout -> LDS -> A-fragment layout (wave-private region)
#pragma unroll
    for (int ni = 0; ni < 4; ++ni)
#pragma unroll
      for (int r = 0; r < 8; ++r)
        Ps[wid][half * 8 + r][ni * 16 + ln] = f2bf(s[ni][r]);

    // O += P * V
#pragma unroll
    for (int kk2 = 0; kk2 < 2; ++kk2) {
      bf16x16 af = ld_fragA(&Ps[wid][ln][kk2 * 32 + half * 16]);
#pragma unroll
      for (int f = 0; f < 8; ++f) {
        bf16x16 bfr = ld_fragB(&Vs[f * 16 + ln][kk2 * 32 + half * 16]);
        o[f] = wmma_bf16(af, bfr, o[f]);
      }
    }
    __syncthreads();
  }

#pragma unroll
  for (int off = 1; off < 16; off <<= 1)
#pragma unroll
    for (int r = 0; r < 8; ++r) lrow[r] += __shfl_xor(lrow[r], off, 32);

#pragma unroll
  for (int f = 0; f < 8; ++f) {
    int hd = f * 16 + ln;
#pragma unroll
    for (int r = 0; r < 8; ++r) {
      int t = qt * 128 + wid * 16 + half * 8 + r;
      y[((size_t)(bb * T_ + t)) * D_ + hh * HD_ + hd] = f2bf(o[f][r] / lrow[r]);
    }
  }
}

// ---------------- Output projection: Y bf16 (8192x2048) x Wo^T -> fp32 ---------
__global__ __launch_bounds__(256) void out_gemm(const __bf16* __restrict__ a,
                                                const __bf16* __restrict__ wb,
                                                float* __restrict__ out) {
  const int K = D_, N = D_;
  __shared__ __bf16 As[2][128][48];
  __shared__ __bf16 Bs[2][128][48];
  const int tid  = threadIdx.x;
  const int lane = tid & 31, wid = tid >> 5;
  const int half = lane >> 4, ln = lane & 15;
  const int wm = wid & 3, wn = wid >> 2;
  const int m0 = blockIdx.y * 128, n0 = blockIdx.x * 128;
  const int waveRow = wm * 32, waveCol = wn * 64;
  const int NK = K / 32;

  f32x8 acc[2][4] = {};

  if (wid == 0) {
    tdm_load_2d(LDS_OFF(&As[0][0][0]), a  + (size_t)m0 * K, K, B_ * T_, 32, 128, K, TDM_CFG_GEMM);
    tdm_load_2d(LDS_OFF(&Bs[0][0][0]), wb + (size_t)n0 * K, K, D_,      32, 128, K, TDM_CFG_GEMM);
  }
  for (int kt = 0; kt < NK; ++kt) {
    const int buf = kt & 1;
    if (wid == 0) {
      if (kt + 1 < NK) {
        const int k1 = (kt + 1) * 32;
        tdm_load_2d(LDS_OFF(&As[buf ^ 1][0][0]), a  + (size_t)m0 * K + k1, K, B_ * T_, 32, 128, K, TDM_CFG_GEMM);
        tdm_load_2d(LDS_OFF(&Bs[buf ^ 1][0][0]), wb + (size_t)n0 * K + k1, K, D_,      32, 128, K, TDM_CFG_GEMM);
        __builtin_amdgcn_s_wait_tensorcnt(2);
      } else {
        __builtin_amdgcn_s_wait_tensorcnt(0);
      }
    }
    __syncthreads();
    bf16x16 af[2], bfr[4];
#pragma unroll
    for (int mi = 0; mi < 2; ++mi)
      af[mi] = ld_fragA(&As[buf][waveRow + mi * 16 + ln][half * 16]);
#pragma unroll
    for (int ni = 0; ni < 4; ++ni)
      bfr[ni] = ld_fragB(&Bs[buf][waveCol + ni * 16 + ln][half * 16]);
#pragma unroll
    for (int mi = 0; mi < 2; ++mi)
#pragma unroll
      for (int ni = 0; ni < 4; ++ni)
        acc[mi][ni] = wmma_bf16(af[mi], bfr[ni], acc[mi][ni]);
    __syncthreads();
  }

#pragma unroll
  for (int mi = 0; mi < 2; ++mi)
#pragma unroll
    for (int ni = 0; ni < 4; ++ni) {
      int col = n0 + waveCol + ni * 16 + ln;
#pragma unroll
      for (int r = 0; r < 8; ++r) {
        int row = m0 + waveRow + mi * 16 + half * 8 + r;
        out[(size_t)row * N + col] = acc[mi][ni][r];
      }
    }
}

// -------------------------------- launcher -------------------------------------
extern "C" void kernel_launch(void* const* d_in, const int* in_sizes, int n_in,
                              void* d_out, int out_size, void* d_ws, size_t ws_size,
                              hipStream_t stream) {
  const float* x     = (const float*)d_in[0];
  const int*   mask  = (const int*)d_in[1];
  const float* w_qkv = (const float*)d_in[2];
  const float* w_o   = (const float*)d_in[3];
  float*       out   = (float*)d_out;

  const size_t nX  = (size_t)B_ * T_ * D_;       // 16.78M
  const size_t nWq = (size_t)3 * D_ * D_;        // 12.58M
  const size_t nWo = (size_t)D_ * D_;            // 4.19M
  const size_t headElems = (size_t)B_ * H_ * T_ * HD_;

  __bf16* xb  = (__bf16*)d_ws;                   // 32MB; reused as Y after k1
  __bf16* wqb = xb + nX;                         // 24MB
  __bf16* wob = wqb + nWq;                       //  8MB
  __bf16* qw  = wob + nWo;                       // 32MB
  __bf16* kw  = qw + headElems;                  // 32MB
  __bf16* vw  = kw + headElems;                  // 32MB
  __bf16* yw  = xb;                              // alias: x dead after qkv_gemm

  cvt_bf16<<<dim3((unsigned)(nX  / 4 / 256)), 256, 0, stream>>>(x,     xb,  (int)(nX  / 4));
  cvt_bf16<<<dim3((unsigned)(nWq / 4 / 256)), 256, 0, stream>>>(w_qkv, wqb, (int)(nWq / 4));
  cvt_bf16<<<dim3((unsigned)(nWo / 4 / 256)), 256, 0, stream>>>(w_o,   wob, (int)(nWo / 4));

  qkv_gemm<<<dim3(3 * D_ / 128, (B_ * T_) / 128), 256, 0, stream>>>(xb, wqb, qw, kw, vw);
  flash_attn<<<dim3(T_ / 128, H_, B_), 256, 0, stream>>>(qw, kw, vw, mask, yw);
  out_gemm<<<dim3(D_ / 128, (B_ * T_) / 128), 256, 0, stream>>>(yw, wob, out);
}
